// CFConv_89567247990894
// MI455X (gfx1250) — matrix operand order repaired and checked by reading, not compiled, not run on hardware
//
#include <hip/hip_runtime.h>
#include <math.h>

// ---------------------------------------------------------------------------
// CFConv (SchNet continuous-filter conv) for MI455X / gfx1250.
// GEMMs on v_wmma_f32_16x16x32_f16 (fp16 in, fp32 accumulate).
// Edge gather/scatter done row-at-a-time: coalesced global_load_b128 gathers
// from L2-resident x_proj and contiguous global_atomic_add_f32 scatters.
// ---------------------------------------------------------------------------

typedef __attribute__((ext_vector_type(16))) _Float16 v16h;
typedef __attribute__((ext_vector_type(8)))  _Float16 v8h;
typedef __attribute__((ext_vector_type(8)))  float    v8f;

#define DIM        128
#define LDS_STRIDE 136   // halves per row: 128 + 8 pad (rotates LDS banks)
#define FSTRIDE    132   // floats per filter row: 128 + 4 pad

__device__ __forceinline__ int lane_id() { return threadIdx.x & 31; }
__device__ __forceinline__ int wave_id() { return threadIdx.x >> 5; }

// ---- WMMA wrapper ---------------------------------------------------------
__device__ __forceinline__ v8f wmma_f16(v16h a, v16h b, v8f c) {
  return __builtin_amdgcn_wmma_f32_16x16x32_f16(false, a, false, b,
                                                (short)0, c, false, false);
}

// ---- A fragment (16x32 fp16, row-major in LDS, stride LDS_STRIDE halves) --
// Lane L holds row M=L%16. Lanes 0-15: K = kbase+{0..15};
// lanes 16-31: K = kbase+{16..31}.
__device__ __forceinline__ v16h load_a_lds(const _Float16* tile, int kbase) {
  const int L     = lane_id();
  const int row   = L & 15;
  const int khalf = (L >> 4) * 16;
  const _Float16* p = tile + row * LDS_STRIDE + kbase + khalf;
  v8h lo = *(const v8h*)p;         // ds_load_b128
  v8h hi = *(const v8h*)(p + 8);   // ds_load_b128
  return __builtin_shufflevector(lo, hi, 0,1,2,3,4,5,6,7,8,9,10,11,12,13,14,15);
}

// ---- B fragment (32x16 fp16) from pre-swizzled weights --------------------
// Tile (ktile,ntile) stored as [lane][16 halves]; lane L = col N=L%16,
// K = (L/16)*16 + h.  Two global_load_b128 per fragment.
__device__ __forceinline__ v16h load_b_swz(const _Float16* w, int ktile, int ntile) {
  const _Float16* p = w + (((ktile << 3) + ntile) << 9) + (lane_id() << 4);
  v8h lo = *(const v8h*)p;
  v8h hi = *(const v8h*)(p + 8);
  return __builtin_shufflevector(lo, hi, 0,1,2,3,4,5,6,7,8,9,10,11,12,13,14,15);
}

// ---- weight pre-swizzle (fp32 row-major [K][128] -> fragment layout) ------
__device__ __forceinline__ int swz_idx(int k, int n) {
  const int ktile = k >> 5, ntile = n >> 4, kk = k & 31, nn = n & 15;
  const int lane = ((kk >> 4) << 4) | nn;
  return (((ktile << 3) + ntile) << 9) + (lane << 4) + (kk & 15);
}

__global__ void prep_weights(const float* __restrict__ W1, const float* __restrict__ Wf1,
                             const float* __restrict__ bf1,
                             const float* __restrict__ Wf2, const float* __restrict__ W2,
                             _Float16* __restrict__ W1h, _Float16* __restrict__ Wf1h,
                             _Float16* __restrict__ Wf2h, _Float16* __restrict__ W2h) {
  const int i = blockIdx.x * blockDim.x + threadIdx.x;
  if (i >= 128 * DIM) return;
  const int k = i >> 7, n = i & 127;
  const int d = swz_idx(k, n);
  W1h[d]  = (_Float16)W1[i];
  Wf2h[d] = (_Float16)Wf2[i];
  W2h[d]  = (_Float16)W2[i];
  if (k < 32) {
    // Wf1 is [16,128]; row 16 carries bf1 (bias folded into K), rows 17..31 = 0
    float v = (k < 16) ? Wf1[k * DIM + n] : ((k == 16) ? bf1[n] : 0.0f);
    Wf1h[d] = (_Float16)v;
  }
}

__global__ void zero_f32(float* __restrict__ p, long n4) {  // n4 = count of float4
  const long stride = (long)gridDim.x * blockDim.x;
  const float4 z = {0.0f, 0.0f, 0.0f, 0.0f};
  for (long i = (long)blockIdx.x * blockDim.x + threadIdx.x; i < n4; i += stride)
    ((float4*)p)[i] = z;
}

// ---------------------------------------------------------------------------
// dst[16-row tile] = src_tile @ W (+bias).  One wave per 16-row tile,
// 8 waves per block.  Safe for src == dst (tile-local in-place).
// ---------------------------------------------------------------------------
__global__ void __launch_bounds__(256)
node_gemm16(const float* __restrict__ src, float* __restrict__ dst,
            const _Float16* __restrict__ Wh, const float* __restrict__ bias,
            int n_tiles) {
  __shared__ _Float16 lds[8][16 * LDS_STRIDE];
  const int tile = blockIdx.x * (blockDim.x >> 5) + wave_id();
  if (tile >= n_tiles) return;
  _Float16* t = lds[wave_id()];
  const int L = lane_id();
  const int rows = tile << 4;

  // stage 16x128 fp32 -> fp16 LDS (float4 loads, 16 per lane)
  for (int idx = L; idx < 16 * 32; idx += 32) {
    const int r = idx >> 5, c4 = (idx & 31) << 2;
    const float4 v = *(const float4*)(src + (long)(rows + r) * DIM + c4);
    _Float16* q = t + r * LDS_STRIDE + c4;
    q[0] = (_Float16)v.x; q[1] = (_Float16)v.y;
    q[2] = (_Float16)v.z; q[3] = (_Float16)v.w;
  }

  v16h a[4];
#pragma unroll
  for (int k = 0; k < 4; ++k) a[k] = load_a_lds(t, k * 32);

  const int cl = L & 15, rbase = (L >> 4) << 3;
#pragma unroll
  for (int nt = 0; nt < 8; ++nt) {
    const float b = bias[nt * 16 + cl];
    v8f c;
#pragma unroll
    for (int i = 0; i < 8; ++i) c[i] = b;
#pragma unroll
    for (int k = 0; k < 4; ++k) c = wmma_f16(a[k], load_b_swz(Wh, k, nt), c);
#pragma unroll
    for (int v = 0; v < 8; ++v)
      dst[(long)(rows + rbase + v) * DIM + nt * 16 + cl] = c[v];
  }
}

// ---------------------------------------------------------------------------
// Per 16-edge tile (one wave, 4 waves/block):
//   filter = silu([rbf | 1] @ [Wf1 ; bf1]) @ Wf2 + bf2     (WMMA)
// then per edge: coalesced row gather of x_proj[col[e]], modulate with the
// LDS-staged filter row, contiguous atomic scatter-add into out[row[e]].
// ---------------------------------------------------------------------------
__global__ void __launch_bounds__(128)
edge_kernel(const float* __restrict__ rbf, const int* __restrict__ ei,
            const float* __restrict__ x_proj,
            const _Float16* __restrict__ Wf1h,
            const _Float16* __restrict__ Wf2h, const float* __restrict__ bf2,
            float* __restrict__ out, int E, int e_tiles) {
  __shared__ _Float16 hls[4][16 * LDS_STRIDE];   // 17408 B
  __shared__ float    fls[4][16 * FSTRIDE];      // 33792 B
  const int tile = blockIdx.x * (blockDim.x >> 5) + wave_id();
  if (tile >= e_tiles) return;
  _Float16* t  = hls[wave_id()];
  float*    fb = fls[wave_id()];
  const int L = lane_id();
  const int ebase = tile << 4;

  // stage rbf [16x16] fp32 -> fp16 cols 0..15, col 16 = 1.0 (bias), 17..31 = 0
  for (int idx = L; idx < 64; idx += 32) {
    const int e = idx >> 2, c4 = (idx & 3) << 2;
    const float4 v = *(const float4*)(rbf + (long)(ebase + e) * 16 + c4);
    _Float16* q = t + e * LDS_STRIDE + c4;
    q[0] = (_Float16)v.x; q[1] = (_Float16)v.y;
    q[2] = (_Float16)v.z; q[3] = (_Float16)v.w;
    _Float16* z = t + e * LDS_STRIDE + 16 + c4;
    z[0] = (_Float16)((c4 == 0) ? 1.0f : 0.0f);
    z[1] = z[2] = z[3] = (_Float16)0.0f;
  }

  const int cl = L & 15, rsel = L >> 4;
  v16h ar = load_a_lds(t, 0);  // rbf fragment captured in registers

  // h = silu([rbf|1] @ [Wf1;bf1]) -> fp16 back into the same LDS tile
#pragma unroll
  for (int nt = 0; nt < 8; ++nt) {
    v8f c;
#pragma unroll
    for (int i = 0; i < 8; ++i) c[i] = 0.0f;
    c = wmma_f16(ar, load_b_swz(Wf1h, 0, nt), c);
#pragma unroll
    for (int v = 0; v < 8; ++v) {
      float hv = c[v];
      hv = hv / (1.0f + __expf(-hv));           // silu
      t[(v + (rsel << 3)) * LDS_STRIDE + nt * 16 + cl] = (_Float16)hv;
    }
  }

  v16h a[4];
#pragma unroll
  for (int k = 0; k < 4; ++k) a[k] = load_a_lds(t, k * 32);

  // filter = h @ Wf2 + bf2 -> fp32 LDS rows (row = local edge)
#pragma unroll
  for (int nt = 0; nt < 8; ++nt) {
    const float b = bf2[nt * 16 + cl];
    v8f c;
#pragma unroll
    for (int i = 0; i < 8; ++i) c[i] = b;
#pragma unroll
    for (int k = 0; k < 4; ++k) c = wmma_f16(a[k], load_b_swz(Wf2h, k, nt), c);
#pragma unroll
    for (int v = 0; v < 8; ++v)
      fb[(v + (rsel << 3)) * FSTRIDE + nt * 16 + cl] = c[v];
  }

  // per-edge: wave-coalesced gather, modulate, contiguous atomic scatter-add
  const int col4 = L << 2;   // this lane's 4 columns
#pragma unroll
  for (int e = 0; e < 16; ++e) {
    const int ge  = ebase + e;
    const int dst = __builtin_amdgcn_readfirstlane(ei[ge]);      // segment id
    const int src = __builtin_amdgcn_readfirstlane(ei[E + ge]);  // gather row
    const float4 xv = *(const float4*)(x_proj + (long)src * DIM + col4);
    const float4 fv = *(const float4*)(fb + e * FSTRIDE + col4);
    float* o = out + (long)dst * DIM + col4;
    atomicAdd(o + 0, xv.x * fv.x);
    atomicAdd(o + 1, xv.y * fv.y);
    atomicAdd(o + 2, xv.z * fv.z);
    atomicAdd(o + 3, xv.w * fv.w);
  }
}

// ---------------------------------------------------------------------------
extern "C" void kernel_launch(void* const* d_in, const int* in_sizes, int n_in,
                              void* d_out, int out_size, void* d_ws, size_t ws_size,
                              hipStream_t stream) {
  const float* x    = (const float*)d_in[0];
  const int*   ei   = (const int*)  d_in[1];   // [2,E] (row, col)
  const float* rbf  = (const float*)d_in[2];   // [E,16]
  const float* W1   = (const float*)d_in[3];
  const float* b1   = (const float*)d_in[4];
  const float* Wf1  = (const float*)d_in[5];
  const float* bf1  = (const float*)d_in[6];
  const float* Wf2  = (const float*)d_in[7];
  const float* bf2  = (const float*)d_in[8];
  const float* W2   = (const float*)d_in[9];
  const float* b2   = (const float*)d_in[10];
  float* out = (float*)d_out;

  const int N = in_sizes[0] / DIM;
  const int E = in_sizes[2] / 16;

  // workspace layout
  char* ws = (char*)d_ws;
  _Float16* W1h  = (_Float16*)(ws);                      // 128x128 fp16 = 32768 B
  _Float16* Wf1h = (_Float16*)(ws + 32768);              //  32x128 fp16 =  8192 B
  _Float16* Wf2h = (_Float16*)(ws + 40960);              // 128x128 fp16 = 32768 B
  _Float16* W2h  = (_Float16*)(ws + 73728);              // 128x128 fp16 = 32768 B
  float*    x_proj = (float*)(ws + 106496);              // N*128 fp32

  const int n_tiles = (N + 15) >> 4;   // 6250
  const int e_tiles = (E + 15) >> 4;   // 40000

  prep_weights<<<(128 * DIM + 255) / 256, 256, 0, stream>>>(
      W1, Wf1, bf1, Wf2, W2, W1h, Wf1h, Wf2h, W2h);

  zero_f32<<<2048, 256, 0, stream>>>(out, ((long)N * DIM) >> 2);

  node_gemm16<<<(n_tiles + 7) / 8, 256, 0, stream>>>(
      x, x_proj, W1h, b1, n_tiles);                      // lin1: x @ W1 + b1

  edge_kernel<<<(e_tiles + 3) / 4, 128, 0, stream>>>(
      rbf, ei, x_proj, Wf1h, Wf2h, bf2, out, E, e_tiles);

  node_gemm16<<<(n_tiles + 7) / 8, 256, 0, stream>>>(
      out, out, W2h, b2, n_tiles);                       // lin2, in-place per tile
}
